// _Attention_62113817035195
// MI455X (gfx1250) — matrix operand order repaired and checked by reading, not compiled, not run on hardware
//
#include <hip/hip_runtime.h>
#include <hip/hip_bf16.h>

typedef __attribute__((ext_vector_type(16))) __bf16 v16bf;
typedef __attribute__((ext_vector_type(8)))  float  v8f;

#define NSPLIT 4
// 1/sqrt(128) * log2(e)
#define SCL 0.12752519897457589f

// A fragment (16xK rows, contiguous-k order, same as load_fragB): lane holds contiguous K elems {2*hi8..2*hi8+15}
static __device__ __forceinline__ v16bf load_fragA(const float* __restrict__ rowk, int hi8) {
  v16bf r;
#pragma unroll
  for (int i = 0; i < 8; ++i) r[i] = (__bf16)rowk[2 * hi8 + i];
#pragma unroll
  for (int i = 0; i < 8; ++i) r[8 + i] = (__bf16)rowk[2 * hi8 + 8 + i];
  return r;
}
// B fragment (Kx16 cols, same k order as load_fragA): lane half holds contiguous K {hi16..hi16+15}
static __device__ __forceinline__ v16bf load_fragB(const float* __restrict__ rowk, int hi16) {
  v16bf r;
#pragma unroll
  for (int i = 0; i < 16; ++i) r[i] = (__bf16)rowk[hi16 + i];
  return r;
}

// ---------------- QKV projection: Y = X @ W^T + b, scattered to (B,H,S,D) ----------------
__global__ __launch_bounds__(32) void qkv_gemm(
    const float* __restrict__ X,
    const float* __restrict__ Wq, const float* __restrict__ bq,
    const float* __restrict__ Wk, const float* __restrict__ bk,
    const float* __restrict__ Wv, const float* __restrict__ bv,
    float* __restrict__ q_ws, float* __restrict__ k_new, float* __restrict__ v_new)
{
  const int ntile = blockIdx.x;   // 0..127
  const int mtile = blockIdx.y;   // 0..7
  const int which = blockIdx.z;   // 0..2
  const float* W    = which == 0 ? Wq : (which == 1 ? Wk : Wv);
  const float* bias = which == 0 ? bq : (which == 1 ? bk : bv);
  float*       dst  = which == 0 ? q_ws : (which == 1 ? k_new : v_new);

  const int lane = threadIdx.x & 31;
  const int ln   = lane & 15;
  const int hi8  = (lane >> 4) << 3;
  const int hi16 = (lane >> 4) << 4;

  const float* xrow = X + (size_t)(mtile * 16 + ln) * 2048;
  const float* wrow = W + (size_t)(ntile * 16 + ln) * 2048;

  v8f acc = {};
  for (int k = 0; k < 2048; k += 32) {
    v16bf a = load_fragA(xrow + k, hi8);
    v16bf b = load_fragB(wrow + k, hi16);
    acc = __builtin_amdgcn_wmma_f32_16x16x32_bf16(false, a, false, b, (short)0, acc, false, false);
  }
  const int n = ntile * 16 + ln;
  const float bval = bias[n];
  const int h = n >> 7, d = n & 127;
#pragma unroll
  for (int v = 0; v < 8; ++v) {
    int m = mtile * 16 + hi8 + v;
    int bb = m >> 4, s = m & 15;
    dst[(((size_t)bb * 16 + h) * 16 + s) * 128 + d] = acc[v] + bval;
  }
}

// ---------------- Flash attention partials ----------------
__global__ __launch_bounds__(256) void attn_partial(
    const float* __restrict__ q_ws,
    const float* __restrict__ k_cache, const float* __restrict__ v_cache,
    const float* __restrict__ k_new,   const float* __restrict__ v_new,
    float* __restrict__ part)
{
  const int split = blockIdx.x;   // 0..3
  const int bh    = blockIdx.y;   // 0..127
  const int tid   = threadIdx.x;
  const int wave  = tid >> 5;
  const int lane  = tid & 31;
  const int ln    = lane & 15;
  const int hi8   = (lane >> 4) << 3;
  const int hi16  = (lane >> 4) << 4;
  const int worker = split * 8 + wave;   // 0..31

  // q^T B-fragments (query s = ln is this lane's column), scale folded in (log2 domain)
  const float* qrow = q_ws + ((size_t)bh * 16 + ln) * 128;
  v16bf qf[4];
#pragma unroll
  for (int kk = 0; kk < 4; ++kk)
#pragma unroll
    for (int i = 0; i < 16; ++i)
      qf[kk][i] = (__bf16)(qrow[kk * 32 + hi16 + i] * SCL);

  const float* kc = k_cache + (size_t)bh * 4096 * 128;
  const float* vc = v_cache + (size_t)bh * 4096 * 128;
  const float* kn = k_new   + (size_t)bh * 16 * 128;
  const float* vn = v_new   + (size_t)bh * 16 * 128;

  float mrow = -3.0e38f, lrow = 0.0f;
  v8f zero = {};
  v8f ctx[8];
#pragma unroll
  for (int i = 0; i < 8; ++i) ctx[i] = zero;

  for (int u = worker; u < 129; u += 32) {     // 129 units of 32 keys cover T=4112
    const int t0 = u * 32;
    const int tA = t0 + ln;
    int tB = t0 + 16 + ln; if (tB > 4111) tB = 4111;
    const float* krA = (tA < 4096) ? kc + (size_t)tA * 128 : kn + (size_t)(tA - 4096) * 128;
    const float* krB = (tB < 4096) ? kc + (size_t)tB * 128 : kn + (size_t)(tB - 4096) * 128;

    v8f st0 = zero, st1 = zero;
#pragma unroll
    for (int kk = 0; kk < 4; ++kk) {
      v16bf a0 = load_fragA(krA + kk * 32, hi8);
      v16bf a1 = load_fragA(krB + kk * 32, hi8);
      st0 = __builtin_amdgcn_wmma_f32_16x16x32_bf16(false, a0, false, qf[kk], (short)0, st0, false, false);
      st1 = __builtin_amdgcn_wmma_f32_16x16x32_bf16(false, a1, false, qf[kk], (short)0, st1, false, false);
    }
    if (t0 + 31 > 4096) {   // only final unit: causal mask t <= s + 4096
#pragma unroll
      for (int v = 0; v < 8; ++v) {
        if (t0 + hi8 + v      > 4096 + ln) st0[v] = -1.0e30f;
        if (t0 + 16 + hi8 + v > 4096 + ln) st1[v] = -1.0e30f;
      }
    }
    // online softmax (scores already in log2 domain)
    float cmax = -3.0e38f;
#pragma unroll
    for (int v = 0; v < 8; ++v) cmax = fmaxf(cmax, fmaxf(st0[v], st1[v]));
    cmax = fmaxf(cmax, __shfl_xor(cmax, 16, 32));
    float mnew  = fmaxf(mrow, cmax);
    float alpha = exp2f(mrow - mnew);
    float p0[8], p1[8], ls = 0.0f;
#pragma unroll
    for (int v = 0; v < 8; ++v) {
      p0[v] = exp2f(st0[v] - mnew);
      p1[v] = exp2f(st1[v] - mnew);
      ls += p0[v] + p1[v];
    }
    ls += __shfl_xor(ls, 16, 32);
    lrow = lrow * alpha + ls;
    mrow = mnew;

    // Build P as B-fragment (K=t contiguous per half-wave) from D-layout score tiles.
    v16bf pf;
    const bool hiHalf = lane >= 16;
#pragma unroll
    for (int e = 0; e < 8; ++e) {
      float o0 = __shfl_xor(p0[e], 16, 32);
      float o1 = __shfl_xor(p1[e], 16, 32);
      pf[e]     = (__bf16)(hiHalf ? o1 : p0[e]);
      pf[8 + e] = (__bf16)(hiHalf ? p1[e] : o0);
    }
    // rescale accumulated context (per-lane s => uniform alpha)
#pragma unroll
    for (int dt = 0; dt < 8; ++dt)
#pragma unroll
      for (int v = 0; v < 8; ++v) ctx[dt][v] *= alpha;

    // V^T A-fragment row pointers (t per A K-pattern), d_local = ln
    const float* vp[16];
#pragma unroll
    for (int e = 0; e < 16; ++e) {
      int tl = hi16 + e;
      int t = t0 + tl; if (t > 4111) t = 4111;     // masked lanes contribute P=0
      vp[e] = ((t < 4096) ? vc + (size_t)t * 128 : vn + (size_t)(t - 4096) * 128) + ln;
    }
#pragma unroll
    for (int dt = 0; dt < 8; ++dt) {
      v16bf vf;
#pragma unroll
      for (int e = 0; e < 16; ++e) vf[e] = (__bf16)vp[e][dt * 16];
      ctx[dt] = __builtin_amdgcn_wmma_f32_16x16x32_bf16(false, vf, false, pf, (short)0, ctx[dt], false, false);
    }
  }

  // -------- deterministic intra-block merge of 8 waves (log tree via LDS) --------
  __shared__ float lctx[4][16 * 128];
  __shared__ float lds_m[8][16];
  __shared__ float lds_l[8][16];

  if (lane < 16) lds_m[wave][ln] = mrow;
  __syncthreads();
  float mstar = -3.0e38f;
#pragma unroll
  for (int j = 0; j < 8; ++j) mstar = fmaxf(mstar, lds_m[j][ln]);
  float w = exp2f(mrow - mstar);
#pragma unroll
  for (int dt = 0; dt < 8; ++dt)
#pragma unroll
    for (int v = 0; v < 8; ++v) ctx[dt][v] *= w;
  if (lane < 16) lds_l[wave][ln] = lrow * w;

  if (wave >= 4) {
    float* p = &lctx[wave - 4][ln * 128];
#pragma unroll
    for (int dt = 0; dt < 8; ++dt)
#pragma unroll
      for (int v = 0; v < 8; ++v) p[dt * 16 + hi8 + v] = ctx[dt][v];
  }
  __syncthreads();
  if (wave < 4) {
    const float* p = &lctx[wave][ln * 128];
#pragma unroll
    for (int dt = 0; dt < 8; ++dt)
#pragma unroll
      for (int v = 0; v < 8; ++v) ctx[dt][v] += p[dt * 16 + hi8 + v];
  }
  __syncthreads();
  if (wave == 2 || wave == 3) {
    float* p = &lctx[wave - 2][ln * 128];
#pragma unroll
    for (int dt = 0; dt < 8; ++dt)
#pragma unroll
      for (int v = 0; v < 8; ++v) p[dt * 16 + hi8 + v] = ctx[dt][v];
  }
  __syncthreads();
  if (wave < 2) {
    const float* p = &lctx[wave][ln * 128];
#pragma unroll
    for (int dt = 0; dt < 8; ++dt)
#pragma unroll
      for (int v = 0; v < 8; ++v) ctx[dt][v] += p[dt * 16 + hi8 + v];
  }
  __syncthreads();
  if (wave == 1) {
    float* p = &lctx[0][ln * 128];
#pragma unroll
    for (int dt = 0; dt < 8; ++dt)
#pragma unroll
      for (int v = 0; v < 8; ++v) p[dt * 16 + hi8 + v] = ctx[dt][v];
  }
  __syncthreads();
  float* pb = part + ((size_t)bh * NSPLIT + split) * 2080;
  if (wave == 0) {
    const float* p = &lctx[0][ln * 128];
#pragma unroll
    for (int dt = 0; dt < 8; ++dt)
#pragma unroll
      for (int v = 0; v < 8; ++v)
        pb[ln * 128 + dt * 16 + hi8 + v] = ctx[dt][v] + p[dt * 16 + hi8 + v];
  }
  if (tid < 16) {
    float lt = 0.0f;
#pragma unroll
    for (int j = 0; j < 8; ++j) lt += lds_l[j][tid];
    pb[2048 + tid] = mstar;
    pb[2064 + tid] = lt;
  }
}

// ---------------- Combine the NSPLIT partials per (b,h); write ctx in (B*S, E) ----------------
__global__ __launch_bounds__(256) void attn_combine(const float* __restrict__ part,
                                                    float* __restrict__ ctx_ws)
{
  const int bh  = blockIdx.x;
  const int tid = threadIdx.x;
  const int s   = tid >> 4;
  const int d0  = (tid & 15) * 8;
  const float* pb = part + (size_t)bh * NSPLIT * 2080;

  float mj[NSPLIT], lj[NSPLIT], mstar = -3.0e38f;
#pragma unroll
  for (int j = 0; j < NSPLIT; ++j) {
    mj[j] = pb[j * 2080 + 2048 + s];
    lj[j] = pb[j * 2080 + 2064 + s];
    mstar = fmaxf(mstar, mj[j]);
  }
  float lt = 0.0f;
#pragma unroll
  for (int j = 0; j < NSPLIT; ++j) lt += lj[j] * exp2f(mj[j] - mstar);
  const float inv = 1.0f / lt;

  float acc[8];
#pragma unroll
  for (int i = 0; i < 8; ++i) acc[i] = 0.0f;
#pragma unroll
  for (int j = 0; j < NSPLIT; ++j) {
    float wj = exp2f(mj[j] - mstar) * inv;
    const float* cj = pb + j * 2080 + s * 128 + d0;
#pragma unroll
    for (int i = 0; i < 8; ++i) acc[i] += wj * cj[i];
  }
  const int b = bh >> 4, h = bh & 15;
  float* orow = ctx_ws + ((size_t)(b * 16 + s)) * 2048 + h * 128 + d0;
#pragma unroll
  for (int i = 0; i < 8; ++i) orow[i] = acc[i];
}

// ---------------- Output projection: out = ctx @ Wo^T + bo (row-major) ----------------
__global__ __launch_bounds__(32) void o_gemm(const float* __restrict__ X,
                                             const float* __restrict__ W,
                                             const float* __restrict__ bias,
                                             float* __restrict__ out)
{
  const int ntile = blockIdx.x;
  const int mtile = blockIdx.y;
  const int lane = threadIdx.x & 31;
  const int ln   = lane & 15;
  const int hi8  = (lane >> 4) << 3;
  const int hi16 = (lane >> 4) << 4;
  const float* xrow = X + (size_t)(mtile * 16 + ln) * 2048;
  const float* wrow = W + (size_t)(ntile * 16 + ln) * 2048;
  v8f acc = {};
  for (int k = 0; k < 2048; k += 32) {
    v16bf a = load_fragA(xrow + k, hi8);
    v16bf b = load_fragB(wrow + k, hi16);
    acc = __builtin_amdgcn_wmma_f32_16x16x32_bf16(false, a, false, b, (short)0, acc, false, false);
  }
  const int n = ntile * 16 + ln;
  const float bval = bias[n];
#pragma unroll
  for (int v = 0; v < 8; ++v)
    out[(size_t)(mtile * 16 + hi8 + v) * 2048 + n] = acc[v] + bval;
}

extern "C" void kernel_launch(void* const* d_in, const int* in_sizes, int n_in,
                              void* d_out, int out_size, void* d_ws, size_t ws_size,
                              hipStream_t stream)
{
  const float* emb    = (const float*)d_in[0];
  const float* kcache = (const float*)d_in[1];
  const float* vcache = (const float*)d_in[2];
  const float* Wq = (const float*)d_in[3]; const float* bq = (const float*)d_in[4];
  const float* Wk = (const float*)d_in[5]; const float* bk = (const float*)d_in[6];
  const float* Wv = (const float*)d_in[7]; const float* bv = (const float*)d_in[8];
  const float* Wo = (const float*)d_in[9]; const float* bo = (const float*)d_in[10];

  float* out   = (float*)d_out;            // (B,S,E) = 262144 floats
  float* k_new = out + 262144;             // (B,H,S,D)
  float* v_new = out + 524288;             // (B,H,S,D)

  float* ws     = (float*)d_ws;
  float* q_ws   = ws;                          // 262144
  float* partb  = ws + 262144;                 // 4*128*2080 = 1064960
  float* ctx_ws = partb + NSPLIT * 128 * 2080; // 262144

  qkv_gemm<<<dim3(128, 8, 3), 32, 0, stream>>>(emb, Wq, bq, Wk, bk, Wv, bv, q_ws, k_new, v_new);
  attn_partial<<<dim3(NSPLIT, 128), 256, 0, stream>>>(q_ws, kcache, vcache, k_new, v_new, partb);
  attn_combine<<<128, 256, 0, stream>>>(partb, ctx_ws);
  o_gemm<<<dim3(128, 8), 32, 0, stream>>>(ctx_ws, Wo, bo, out);
}
